// TensorProduct_60009283060152
// MI455X (gfx1250) — matrix operand order repaired and checked by reading, compiled, not run on hardware
//
#include <hip/hip_runtime.h>

// ============================================================================
// e3nn fully-connected tensor product (128x0+64x1+32x2)^2 -> 128x0+64x1+32x2
// Strategy (MI455X / gfx1250, wave32, WMMA):
//   out[m,w,k] = sum_{i,j} C[k;i,j] * sum_{u,v} W[u,v,w] x1[m,u,i] x2[m,v,j]
//   Inner (u,v)->w contraction is a GEMM: A = outer-product fragments built
//   on the fly from LDS-resident f16 tiles of x1/x2; B = pre-swizzled f16
//   weight fragments (L2-resident); accumulate f32 via v_wmma_f32_16x16x32_f16.
// ============================================================================

typedef __attribute__((ext_vector_type(16))) _Float16 v16h;
typedef __attribute__((ext_vector_type(8)))  float    v8f;

namespace tp {

constexpr int M_BATCH = 8192;
constexpr int DIM     = 480;   // 128*1 + 64*3 + 32*5
constexpr int NROWS   = 32;    // batch rows per workgroup (2 WMMA M-tiles)
constexpr int BLOCK   = 448;   // 14 waves of 32
constexpr int DIMP    = 488;   // padded LDS row stride (halves); 488/2=244 dwords,
                               // 244 mod 64 = 52 -> 16-lane b128 reads tile all 64 banks

constexpr int MUL[3]  = {128, 64, 32};
constexpr int DIML[3] = {1, 3, 5};
constexpr int OFFS[3] = {0, 128, 320};

constexpr int NI = 15;
// INSTRUCTIONS in reference order (i1-major, i2, i3):
constexpr int I1T[NI] = {0,0,0,1,1,1,1,1,1,2,2,2,2,2,2};
constexpr int I2T[NI] = {0,1,2,0,1,1,1,2,2,0,1,1,2,2,2};
constexpr int I3T[NI] = {0,1,2,1,0,1,2,1,2,2,1,2,0,1,2};
// coupling type: 0=general(w3j), 1=(0,0,0), 2=l1==0 (delta_kj/sqrt(d3)),
//                3=l2==0 (delta_ki/sqrt(d3)), 4=l3==0 (delta_ij/sqrt(d1))
constexpr int CTT[NI] = {1,2,2,3,4,0,0,0,0,3,0,0,4,0,0};
// per-instruction offsets (in halves) of swizzled f16 weights in workspace
constexpr int WS_OFF_H[NI+1] = {
  0, 2097152, 2621440, 2752512, 3276800, 3801088, 4063232, 4194304,
  4325376, 4390912, 4521984, 4653056, 4718592, 4849664, 4882432, 4915200};
// total: 4,915,200 halves = 9,830,400 bytes of d_ws

struct Params {
  const float* x1;
  const float* x2;
  const float* w[NI];    // weight_{i1}_{i2}_{i3}, shape (mul1, mul2, mul3)
  const float* w3j[NI];  // w3j, shape (d3, d1, d2)
  _Float16*    wsW;      // swizzled f16 weights (B fragments)
  float*       out;
};

// ----------------------------------------------------------------------------
// Prologue: convert weights f32->f16 and swizzle into WMMA B-fragment layout.
// Fragment = 32x16 (KxN) = 512 halves; per lane 16 contiguous halves:
//   lanes 0-15 : column n = lane,    K = kb*32 + 0..15  (2 per VGPR dword)
//   lanes 16-31: column n = lane-16, K = kb*32 + 16..31
// B element (K,n) = Wflat[K*mul3 + n]  since K = u*mul2 + v.
// ----------------------------------------------------------------------------
__global__ void swizzle_w(Params p) {
  const int total = WS_OFF_H[NI] >> 4;         // one thread per 16-half lane chunk
  int tid = blockIdx.x * blockDim.x + threadIdx.x;
  if (tid >= total) return;
  const int h = tid << 4;
  int inst = 0;
#pragma unroll
  for (int t = 1; t < NI; ++t) if (h >= WS_OFF_H[t]) inst = t;
  const int local = h - WS_OFF_H[inst];
  const int mul3  = MUL[I3T[inst]];
  const int nt3   = mul3 >> 4;
  const int frag  = local >> 9;                // 512 halves per fragment
  const int lane  = (local >> 4) & 31;
  const int kb    = frag / nt3;
  const int nt    = frag - kb * nt3;
  const int n     = (nt << 4) + (lane & 15);
  const int kbase = (kb << 5) + ((lane & 16) ? 16 : 0);
  const float* W  = p.w[inst];
  _Float16* dst   = p.wsW + h;
#pragma unroll
  for (int q = 0; q < 16; ++q)
    dst[q] = (_Float16)W[(size_t)(kbase + q) * mul3 + n];
}

// ----------------------------------------------------------------------------
// Per-wave body, templated on output irrep so d3 is compile-time (keeps the
// f32 accumulators in VGPRs, fully unrolled k loops).
// ----------------------------------------------------------------------------
template <int I3>
__device__ __forceinline__ void wave_body(const Params& p,
                                          const _Float16* __restrict__ sx1,
                                          const _Float16* __restrict__ sx2,
                                          int rowbase, int wt, int lane) {
  constexpr int d3    = 2 * I3 + 1;
  constexpr int mul3  = MUL[I3];
  constexpr int nt3   = mul3 >> 4;
  constexpr int base3 = OFFS[I3];
  constexpr float RSQ3 = 0.57735026918962576f;  // 1/sqrt(3)
  constexpr float RSQ5 = 0.44721359549995794f;  // 1/sqrt(5)

  const int row0 = lane & 15;        // A-row / D-column index within tile
  const int hi8  = (lane >> 4) << 3; // 0 / 8 : K-half select and D-row offset

  v8f acc[2][d3];
  const v8f vzero = {};
#pragma unroll
  for (int mt = 0; mt < 2; ++mt)
#pragma unroll
    for (int k = 0; k < d3; ++k) acc[mt][k] = vzero;

#pragma unroll
  for (int t = 0; t < NI; ++t) {
    if (I3T[t] != I3) continue;                    // folds at compile time
    const int d1   = DIML[I1T[t]];
    const int d2   = DIML[I2T[t]];
    const int mul1 = MUL[I1T[t]];
    const int mul2 = MUL[I2T[t]];
    const int off1 = OFFS[I1T[t]];
    const int off2 = OFFS[I2T[t]];
    const int ct   = CTT[t];
    const float nrm = (ct == 4) ? ((d1 == 3) ? RSQ3 : RSQ5)
                                : ((d3 == 3) ? RSQ3 : (d3 == 5 ? RSQ5 : 1.0f));
    const float* __restrict__ w3 = p.w3j[t];
    const _Float16* __restrict__ wsW = p.wsW + WS_OFF_H[t];
    const int nvb = mul2 >> 5;                     // 32-wide K slabs per u

    for (int i = 0; i < d1; ++i) {
      for (int j = 0; j < d2; ++j) {
        // coupling column C[0..d3-1; i, j]
        float ck[d3];
        float cmax = 0.f;
#pragma unroll
        for (int k = 0; k < d3; ++k) {
          float c;
          if (ct == 1)      c = 1.0f;
          else if (ct == 2) c = (k == j) ? nrm : 0.0f;
          else if (ct == 3) c = (k == i) ? nrm : 0.0f;
          else if (ct == 4) c = (k == 0 && i == j) ? nrm : 0.0f;
          else              c = w3[(k * d1 + i) * d2 + j];
          ck[k] = c;
          cmax = fmaxf(cmax, fabsf(c));
        }
        // wave-uniform skip of structurally-zero couplings (EXEC stays all-1s)
        if (!__builtin_amdgcn_readfirstlane(cmax > 1e-7f ? 1 : 0)) continue;

        v8f g0 = vzero, g1 = vzero;                // per-pair GEMM accumulators
        const int x1col = off1 + i * mul1;
        const int x2col = off2 + j * mul2 + (hi8 << 1);

        for (int u = 0; u < mul1; ++u) {
          // x1 scalar is K-block invariant: hoisted out of the v loop
          const _Float16 s1a = sx1[row0 * DIMP + x1col + u];
          const _Float16 s1b = sx1[(16 + row0) * DIMP + x1col + u];
          for (int vbi = 0; vbi < nvb; ++vbi) {
            const int kb = u * nvb + vbi;
            // B: pre-swizzled weight fragment, 32B contiguous per lane
            const v16h b = *(const v16h*)(wsW + (((kb * nt3 + wt) << 9) + (lane << 4)));
            // A (tile 0): outer product x1[m,u,i] * x2[m,vb..vb+31,j]
            const _Float16* x2p = &sx2[row0 * DIMP + x2col + (vbi << 5)];
            v16h a0;
            ((uint4*)&a0)[0] = *(const uint4*)(x2p);
            ((uint4*)&a0)[1] = *(const uint4*)(x2p + 8);
            a0 = a0 * s1a;                          // v_pk_mul_f16 x8
            g0 = __builtin_amdgcn_wmma_f32_16x16x32_f16(
                     false, a0, false, b, (short)0, g0, false, false);
            // A (tile 1): second 16-row tile reuses the same B fragment
            const _Float16* x2q = &sx2[(16 + row0) * DIMP + x2col + (vbi << 5)];
            v16h a1;
            ((uint4*)&a1)[0] = *(const uint4*)(x2q);
            ((uint4*)&a1)[1] = *(const uint4*)(x2q + 8);
            a1 = a1 * s1b;
            g1 = __builtin_amdgcn_wmma_f32_16x16x32_f16(
                     false, a1, false, b, (short)0, g1, false, false);
          }
        }
        // scatter G into the d3 output components with coupling weights
#pragma unroll
        for (int k = 0; k < d3; ++k) {
#pragma unroll
          for (int e = 0; e < 8; ++e) {
            acc[0][k][e] += ck[k] * g0[e];
            acc[1][k][e] += ck[k] * g1[e];
          }
        }
      }
    }
  }

  // Store: D layout — lane%16 = N column, VGPR e = row e (lanes<16) / e+8.
  // Each output element owned by exactly one wave: plain stores, no atomics.
#pragma unroll
  for (int mt = 0; mt < 2; ++mt)
#pragma unroll
    for (int k = 0; k < d3; ++k)
#pragma unroll
      for (int e = 0; e < 8; ++e) {
        const int m = rowbase + mt * 16 + e + hi8;
        const int col = base3 + (wt * 16 + row0) * d3 + k;
        p.out[(size_t)m * DIM + col] = acc[mt][k][e];
      }
}

// ----------------------------------------------------------------------------
// Main kernel: one workgroup per 32 batch rows; 14 waves own the 8/4/2
// output-channel tiles of irreps l=0/1/2.
// ----------------------------------------------------------------------------
__global__ __launch_bounds__(BLOCK) void tp_kernel(Params p) {
  __shared__ _Float16 sx1[NROWS * DIMP];  // 31,232 B
  __shared__ _Float16 sx2[NROWS * DIMP];  // 31,232 B (total 62,464 B LDS)
  const int rowbase = blockIdx.x * NROWS;

  // Stage x1/x2 tiles: f32 -> f16, transposed component-major per irrep
  // (global col = off + u*d + i  ->  LDS col = off + i*mul + u)
  for (int t = threadIdx.x; t < NROWS * DIM; t += BLOCK) {
    const int r = t / DIM;
    const int c = t - r * DIM;
    int tc;
    if (c < 128)      tc = c;
    else if (c < 320) { const int q = c - 128; tc = 128 + (q % 3) * 64 + q / 3; }
    else              { const int q = c - 320; tc = 320 + (q % 5) * 32 + q / 5; }
    const size_t g = (size_t)(rowbase + r) * DIM + c;
    sx1[r * DIMP + tc] = (_Float16)p.x1[g];
    sx2[r * DIMP + tc] = (_Float16)p.x2[g];
  }
  __syncthreads();

  const int lane = threadIdx.x & 31;
  const int wave = threadIdx.x >> 5;
  if (wave < 8)       wave_body<0>(p, sx1, sx2, rowbase, wave,      lane);
  else if (wave < 12) wave_body<1>(p, sx1, sx2, rowbase, wave - 8,  lane);
  else                wave_body<2>(p, sx1, sx2, rowbase, wave - 12, lane);
}

} // namespace tp

extern "C" void kernel_launch(void* const* d_in, const int* in_sizes, int n_in,
                              void* d_out, int out_size, void* d_ws, size_t ws_size,
                              hipStream_t stream) {
  (void)in_sizes; (void)n_in; (void)out_size; (void)ws_size;
  using namespace tp;
  Params p;
  p.x1 = (const float*)d_in[0];
  p.x2 = (const float*)d_in[1];
  for (int t = 0; t < NI; ++t) {
    p.w[t]   = (const float*)d_in[2 + t];        // weights dict, insertion order
    p.w3j[t] = (const float*)d_in[2 + NI + t];   // w3js dict, insertion order
  }
  p.wsW = (_Float16*)d_ws;                       // needs 9,830,400 B of scratch
  p.out = (float*)d_out;

  // 1) f32->f16 convert + swizzle weights into WMMA B-fragment layout
  const int swz_threads = WS_OFF_H[NI] >> 4;     // 307,200
  swizzle_w<<<(swz_threads + 255) / 256, 256, 0, stream>>>(p);

  // 2) fused tensor product: 256 workgroups x 14 waves
  tp_kernel<<<M_BATCH / NROWS, BLOCK, 0, stream>>>(p);
}
